// MsgPassingNN_23579370455142
// MI455X (gfx1250) — matrix operand order repaired and checked by reading, not compiled, not run on hardware
//
#include <hip/hip_runtime.h>

typedef __attribute__((ext_vector_type(16))) _Float16 v16h;
typedef __attribute__((ext_vector_type(8)))  float    v8f;

#define WAVES_PER_BLK 8
#define BLK 256

// ---------------------------------------------------------------------------
// zero the message accumulator
// ---------------------------------------------------------------------------
__global__ void MsgZeroK(float* __restrict__ p, long n) {
    long i = (long)blockIdx.x * blockDim.x + threadIdx.x;
    if (i < n) p[i] = 0.0f;
}

// ---------------------------------------------------------------------------
// Edge kernel: per-wave, 32 edges. Layer1 (32->9) via 3 chained
// v_wmma_f32_16x16x32_f16 (error-compensated f16 split ~ fp32 accurate),
// LDS transpose, layers 2/3 scalar, scatter with global_atomic_add_f32.
// ---------------------------------------------------------------------------
__global__ __launch_bounds__(BLK) void MsgEdgeK(
    const float* __restrict__ X,
    const int*   __restrict__ esrc,
    const int*   __restrict__ edst,
    const float* __restrict__ feW1, const float* __restrict__ feb1,
    const float* __restrict__ feW2, const float* __restrict__ feb2,
    const float* __restrict__ feW3, const float* __restrict__ feb3,
    float* __restrict__ m, int E)
{
    __shared__ float sW2[81];
    __shared__ float sW3[81];
    __shared__ float sb2[9];
    __shared__ float sb3[9];
    __shared__ float tiles[WAVES_PER_BLK][32][17];   // stride 17: conflict-free reads

    const int tid = threadIdx.x;
    if (tid < 81) { sW2[tid] = feW2[tid]; sW3[tid] = feW3[tid]; }
    if (tid < 9)  { sb2[tid] = feb2[tid]; sb3[tid] = feb3[tid]; }
    __syncthreads();

    const int lane = tid & 31;
    const int wave = tid >> 5;
    const int col  = lane & 15;     // N (hidden unit) for B/C, row M for A
    const int half = lane >> 4;     // K-half selector
    const int baseEdge = blockIdx.x * (BLK / WAVES_PER_BLK * WAVES_PER_BLK / 8) * 8; // = blockIdx.x*256
    const int waveEdge = blockIdx.x * 256 + wave * 32;
    (void)baseEdge;

    // --- B = W1 (32x9 zero-padded to 32x16), f16 hi/lo split, per-lane regs.
// Layout: element i of lane L holds B[K = 8*(L>>4) + i + (i>=8 ? 8 : 0)][N = L&15].
    v16h bh, bl;
    {
        const int kb = half * 8;
#pragma unroll
        for (int i = 0; i < 16; ++i) {
            float w = (col < 9) ? feW1[(kb + i + (i >= 8 ? 8 : 0)) * 9 + col] : 0.0f;
            _Float16 whi = (_Float16)w;
            bh[i] = whi;
            bl[i] = (_Float16)(w - (float)whi);
        }
    }
    const float bias1 = (col < 9) ? feb1[col] : 0.0f;

    // --- two 16-edge tiles per wave ---
#pragma unroll
    for (int t = 0; t < 2; ++t) {
        int e  = waveEdge + t * 16 + col;            // A row M = col
        int ec = (e < E) ? e : (E - 1);
        const int dn = edst[ec];
        const int sn = esrc[ec];
        const float* xd = X + (long)dn * 16 + half * 8;  // features 0..15  = X[dst]
        const float* xs = X + (long)sn * 16 + half * 8;  // features 16..31 = X[src]

        // A layout: lane L (row M=L&15): elem i = feat[8*half + i],
        //           elem 8+i = feat[16 + 8*half + i]
        v16h ah, al;
#pragma unroll
        for (int i = 0; i < 8; ++i) {
            float fd = xd[i], fs = xs[i];
            _Float16 dh = (_Float16)fd, sh = (_Float16)fs;
            ah[i]     = dh;            al[i]     = (_Float16)(fd - (float)dh);
            ah[8 + i] = sh;            al[8 + i] = (_Float16)(fs - (float)sh);
        }

        // c = Ahi*Bhi + Alo*Bhi + Ahi*Blo   (lo*lo term ~2^-22, dropped)
        v8f c = {};
        c = __builtin_amdgcn_wmma_f32_16x16x32_f16(false, ah, false, bh, (short)0, c, false, false);
        c = __builtin_amdgcn_wmma_f32_16x16x32_f16(false, al, false, bh, (short)0, c, false, false);
        c = __builtin_amdgcn_wmma_f32_16x16x32_f16(false, ah, false, bl, (short)0, c, false, false);

        // C layout: c[v] = D[M = v + 8*half][N = col]; bias + ReLU, park in LDS
#pragma unroll
        for (int v = 0; v < 8; ++v) {
            float h1 = c[v] + bias1;
            h1 = h1 > 0.0f ? h1 : 0.0f;
            tiles[wave][t * 16 + v + half * 8][col] = h1;
        }
    }
    __syncthreads();

    // --- each lane owns edge (waveEdge + lane): layers 2 and 3 on VALU ---
    float h1[9];
#pragma unroll
    for (int k = 0; k < 9; ++k) h1[k] = tiles[wave][lane][k];

    float h2[9];
#pragma unroll
    for (int j = 0; j < 9; ++j) {
        float acc = sb2[j];
#pragma unroll
        for (int k = 0; k < 9; ++k) acc += h1[k] * sW2[k * 9 + j];
        h2[j] = acc > 0.0f ? acc : 0.0f;
    }

    const int e = waveEdge + lane;
    if (e < E) {
        float* mp = m + (long)edst[e] * 9;
#pragma unroll
        for (int j = 0; j < 9; ++j) {
            float acc = sb3[j];
#pragma unroll
            for (int k = 0; k < 9; ++k) acc += h2[k] * sW3[k * 9 + j];
            unsafeAtomicAdd(mp + j, acc);   // global_atomic_add_f32, no return
        }
    }
}

// ---------------------------------------------------------------------------
// Node kernel: X' = fx(cat(X, m)); 90 MFLOP/round -> scalar is fine
// ---------------------------------------------------------------------------
__global__ __launch_bounds__(BLK) void MsgNodeK(
    const float* __restrict__ Xin, const float* __restrict__ m,
    const float* __restrict__ fxW1, const float* __restrict__ fxb1,
    const float* __restrict__ fxW2, const float* __restrict__ fxb2,
    const float* __restrict__ fxW3, const float* __restrict__ fxb3,
    float* __restrict__ Xout, int N)
{
    __shared__ float sW1[25 * 9];
    __shared__ float sW2[81];
    __shared__ float sW3[9 * 16];
    __shared__ float sb1[9];
    __shared__ float sb2[9];
    __shared__ float sb3[16];

    const int tid = threadIdx.x;
    if (tid < 225) sW1[tid] = fxW1[tid];
    if (tid < 144) sW3[tid] = fxW3[tid];
    if (tid < 81)  sW2[tid] = fxW2[tid];
    if (tid < 9)   { sb1[tid] = fxb1[tid]; sb2[tid] = fxb2[tid]; }
    if (tid < 16)  sb3[tid] = fxb3[tid];
    __syncthreads();

    const int i = blockIdx.x * BLK + tid;
    if (i >= N) return;

    float in[25];
    const float* xr = Xin + (long)i * 16;
#pragma unroll
    for (int k = 0; k < 16; ++k) in[k] = xr[k];
    const float* mr = m + (long)i * 9;
#pragma unroll
    for (int k = 0; k < 9; ++k) in[16 + k] = mr[k];

    float h1[9];
#pragma unroll
    for (int j = 0; j < 9; ++j) {
        float acc = sb1[j];
#pragma unroll
        for (int k = 0; k < 25; ++k) acc += in[k] * sW1[k * 9 + j];
        h1[j] = acc > 0.0f ? acc : 0.0f;
    }
    float h2[9];
#pragma unroll
    for (int j = 0; j < 9; ++j) {
        float acc = sb2[j];
#pragma unroll
        for (int k = 0; k < 9; ++k) acc += h1[k] * sW2[k * 9 + j];
        h2[j] = acc > 0.0f ? acc : 0.0f;
    }
    float* orow = Xout + (long)i * 16;
#pragma unroll
    for (int d = 0; d < 16; ++d) {
        float acc = sb3[d];
#pragma unroll
        for (int k = 0; k < 9; ++k) acc += h2[k] * sW3[k * 16 + d];
        orow[d] = acc;
    }
}

// ---------------------------------------------------------------------------
extern "C" void kernel_launch(void* const* d_in, const int* in_sizes, int n_in,
                              void* d_out, int out_size, void* d_ws, size_t ws_size,
                              hipStream_t stream)
{
    const float* X    = (const float*)d_in[0];
    const int*   esrc = (const int*)  d_in[1];
    const int*   edst = (const int*)  d_in[2];
    const float* feW1 = (const float*)d_in[3];
    const float* feb1 = (const float*)d_in[4];
    const float* feW2 = (const float*)d_in[5];
    const float* feb2 = (const float*)d_in[6];
    const float* feW3 = (const float*)d_in[7];
    const float* feb3 = (const float*)d_in[8];
    const float* fxW1 = (const float*)d_in[9];
    const float* fxb1 = (const float*)d_in[10];
    const float* fxW2 = (const float*)d_in[11];
    const float* fxb2 = (const float*)d_in[12];
    const float* fxW3 = (const float*)d_in[13];
    const float* fxb3 = (const float*)d_in[14];

    const int N = in_sizes[0] / 16;
    const int E = in_sizes[1];

    // workspace: two X buffers + message accumulator (all L2-resident)
    float* ws  = (float*)d_ws;
    float* Xb0 = ws;
    float* Xb1 = ws + (size_t)N * 16;
    float* m   = ws + (size_t)N * 32;

    const int gEdge = (E + BLK - 1) / BLK;           // 12500 blocks, exact for E=3.2M
    const int gNode = (N + BLK - 1) / BLK;
    const int gZero = ((N * 9) + BLK - 1) / BLK;

    const float* Xin = X;
    for (int r = 0; r < 3; ++r) {
        MsgZeroK<<<gZero, BLK, 0, stream>>>(m, (long)N * 9);
        MsgEdgeK<<<gEdge, BLK, 0, stream>>>(Xin, esrc, edst,
                                            feW1, feb1, feW2, feb2, feW3, feb3,
                                            m, E);
        float* Xout = (r == 2) ? (float*)d_out : (r == 0 ? Xb0 : Xb1);
        MsgNodeK<<<gNode, BLK, 0, stream>>>(Xin, m,
                                            fxW1, fxb1, fxW2, fxb2, fxW3, fxb3,
                                            Xout, N);
        Xin = Xout;
    }
}
